// VectorQuantizer_78640851190409
// MI455X (gfx1250) — matrix operand order repaired and checked by reading, not compiled, not run on hardware
//
#include <hip/hip_runtime.h>
#include <hip/hip_bf16.h>

#define BATCH       32768
#define DIM         256
#define NUM_LATENTS 4096

typedef __bf16    v16bf __attribute__((ext_vector_type(16)));
typedef float     v8f   __attribute__((ext_vector_type(8)));
typedef unsigned  v8u   __attribute__((ext_vector_type(8)));

// round-to-nearest-even f32 -> bf16 bits
__device__ __forceinline__ unsigned short f2bf_bits(float f) {
    unsigned u = __builtin_bit_cast(unsigned, f);
    unsigned r = u + 0x7FFFu + ((u >> 16) & 1u);
    return (unsigned short)(r >> 16);
}

// ---------------------------------------------------------------------------
// K0a: codebook row inv-norms; zero counts + sq-error accumulator
// ---------------------------------------------------------------------------
__global__ void vq_prep(const float* __restrict__ cb, float* __restrict__ invn,
                        unsigned* __restrict__ counts, float* __restrict__ sqacc) {
    int n = blockIdx.x * blockDim.x + threadIdx.x;
    if (n < NUM_LATENTS) {
        float s = 0.f;
        for (int i = 0; i < DIM; ++i) { float v = cb[n * DIM + i]; s += v * v; }
        invn[n]   = 1.0f / fmaxf(sqrtf(s), 1e-12f);
        counts[n] = 0u;
    }
    if (blockIdx.x == 0 && threadIdx.x == 0) *sqacc = 0.f;
}

// ---------------------------------------------------------------------------
// K0b: pack normalized codebook into WMMA B-fragment layout.
// dword index g = ((c*8 + t)*32 + L)*8 + j
//   code n = c*16 + (L&15), half = L>>4, K = t*32 + 16*half + 2*j  (pair K,K+1)
// so in the main loop lane L loads 8 consecutive dwords = one B fragment.
// ---------------------------------------------------------------------------
__global__ void vq_pack_b(const float* __restrict__ cb, const float* __restrict__ invn,
                          unsigned* __restrict__ frags) {
    unsigned g    = blockIdx.x * blockDim.x + threadIdx.x; // 524288 total
    unsigned j    = g & 7u;
    unsigned L    = (g >> 3) & 31u;
    unsigned t    = (g >> 8) & 7u;
    unsigned c    = g >> 11;
    unsigned n    = c * 16u + (L & 15u);
    unsigned half = L >> 4;
    unsigned k    = t * 32u + half * 16u + 2u * j;
    float s  = invn[n];
    float f0 = cb[n * DIM + k] * s;
    float f1 = cb[n * DIM + k + 1] * s;
    frags[g] = (unsigned)f2bf_bits(f0) | ((unsigned)f2bf_bits(f1) << 16);
}

// ---------------------------------------------------------------------------
// K1: main WMMA sweep. One wave per block; 32 batch rows per wave
// (two 16-row A tiles) so every B fragment load feeds two WMMAs.
// Accumulators are seeded with C=2.0 (inline constant) so final scores are
// positive floats; argmax runs on packed keys held as v8u vectors:
//   key = (score bits & ~0xFFF) | (4095 - code_index),  winner = max key.
// ---------------------------------------------------------------------------
__global__ __launch_bounds__(32)
__attribute__((amdgpu_waves_per_eu(4)))
void vq_main(
    const float* __restrict__ x, const float* __restrict__ cb,
    const unsigned* __restrict__ frags, unsigned* __restrict__ counts,
    float* __restrict__ sqacc, float* __restrict__ out) {

    __shared__ float xs[32][260];   // padded stride 260 -> conflict-free column reads
    __shared__ float snorm[32];
    __shared__ int   sidx[32];

    const int lane = threadIdx.x & 31;
    const int row0 = blockIdx.x * 32;

    // stage 32x256 x-tile into LDS
    for (int r = 0; r < 32; ++r)
        for (int i = lane; i < DIM; i += 32)
            xs[r][i] = x[(size_t)(row0 + r) * DIM + i];
    __syncthreads();

    // row norms: one row per lane
    {
        float s = 0.f;
        for (int i = 0; i < DIM; ++i) { float v = xs[lane][i]; s += v * v; }
        snorm[lane] = 1.0f / fmaxf(sqrtf(s), 1e-12f);
    }
    __syncthreads();

    // build resident A fragments for both 16-row tiles (contiguous k = 16*half + s, same order as packed B)
    const int m    = lane & 15;
    const int half = lane >> 4;
    v16bf a0[8], a1[8];
    {
        const float in0 = snorm[m];
        const float in1 = snorm[m + 16];
#pragma unroll
        for (int t = 0; t < 8; ++t) {
#pragma unroll
            for (int j = 0; j < 8; ++j) {
                int k = t * 32 + 2 * j + 16 * half;
                a0[t][2 * j]     = __builtin_bit_cast(__bf16, f2bf_bits(xs[m][k] * in0));
                a0[t][2 * j + 1] = __builtin_bit_cast(__bf16, f2bf_bits(xs[m][k + 1] * in0));
                a1[t][2 * j]     = __builtin_bit_cast(__bf16, f2bf_bits(xs[m + 16][k] * in1));
                a1[t][2 * j + 1] = __builtin_bit_cast(__bf16, f2bf_bits(xs[m + 16][k + 1] * in1));
            }
        }
    }

    // C seed: 2.0 keeps all scores positive so unsigned key compare is valid
    const v8f cst2 = {2.f, 2.f, 2.f, 2.f, 2.f, 2.f, 2.f, 2.f};

    v8u key0 = {0u, 0u, 0u, 0u, 0u, 0u, 0u, 0u};
    v8u key1 = {0u, 0u, 0u, 0u, 0u, 0u, 0u, 0u};

    const int      col    = lane & 15;
    const unsigned invcol = 4095u - (unsigned)col;

#pragma unroll 1
    for (int c = 0; c < NUM_LATENTS / 16; ++c) {
        v16bf b = *(const v16bf*)(frags + c * 2048 + lane * 8);
        v8f acc0 = __builtin_amdgcn_wmma_f32_16x16x32_bf16(
            false, a0[0], false, b, (short)0, cst2, false, false);
        v8f acc1 = __builtin_amdgcn_wmma_f32_16x16x32_bf16(
            false, a1[0], false, b, (short)0, cst2, false, false);
#pragma unroll
        for (int t = 1; t < 8; ++t) {
            b = *(const v16bf*)(frags + c * 2048 + t * 256 + lane * 8);
            acc0 = __builtin_amdgcn_wmma_f32_16x16x32_bf16(
                false, a0[t], false, b, (short)0, acc0, false, false);
            acc1 = __builtin_amdgcn_wmma_f32_16x16x32_bf16(
                false, a1[t], false, b, (short)0, acc1, false, false);
        }
        const unsigned invc = invcol - (unsigned)(c * 16);
        v8u k0 = (__builtin_bit_cast(v8u, acc0) & 0xFFFFF000u) | invc;
        v8u k1 = (__builtin_bit_cast(v8u, acc1) & 0xFFFFF000u) | invc;
        key0 = __builtin_elementwise_max(key0, k0);
        key1 = __builtin_elementwise_max(key1, k1);
    }

    // reduce keys within each 16-lane group; max key = best score, smallest index
    unsigned kk0[8], kk1[8];
#pragma unroll
    for (int r = 0; r < 8; ++r) {
        unsigned k0 = key0[r], k1 = key1[r];
#pragma unroll
        for (int sft = 1; sft < 16; sft <<= 1) {
            unsigned o0 = (unsigned)__shfl_xor((int)k0, sft, 16);
            if (o0 > k0) k0 = o0;
            unsigned o1 = (unsigned)__shfl_xor((int)k1, sft, 16);
            if (o1 > k1) k1 = o1;
        }
        kk0[r] = k0; kk1[r] = k1;
    }

    if (col == 0) {
#pragma unroll
        for (int r = 0; r < 8; ++r) {
            int m0 = r + 8 * half;            // tile0 rows 0..15
            int m1 = 16 + r + 8 * half;       // tile1 rows 16..31
            int i0 = 4095 - (int)(kk0[r] & 0xFFFu);
            int i1 = 4095 - (int)(kk1[r] & 0xFFFu);
            sidx[m0] = i0;
            sidx[m1] = i1;
            atomicAdd(&counts[i0], 1u);
            atomicAdd(&counts[i1], 1u);
            out[(size_t)BATCH * DIM + 5 + row0 + m0] = (float)i0;
            out[(size_t)BATCH * DIM + 5 + row0 + m1] = (float)i1;
        }
    }
    __syncthreads();

    // gather z (unnormalized codebook), write z_q = x + (z - x), accumulate (x-z)^2
    float sq = 0.f;
    for (int r = 0; r < 32; ++r) {
        const float* zr = cb + (size_t)sidx[r] * DIM;
        for (int i = lane; i < DIM; i += 32) {
            float zv = zr[i];
            float xv = xs[r][i];
            float d  = xv - zv;
            sq += d * d;
            out[(size_t)(row0 + r) * DIM + i] = xv + (zv - xv);
        }
    }
#pragma unroll
    for (int sft = 1; sft < 32; sft <<= 1) sq += __shfl_xor(sq, sft);
    if (lane == 0) atomicAdd(sqacc, sq);
}

// ---------------------------------------------------------------------------
// K2: entropy + scalar losses
// ---------------------------------------------------------------------------
__global__ void vq_final(const unsigned* __restrict__ counts,
                         const float* __restrict__ sqacc, float* __restrict__ out) {
    __shared__ float sh[256];
    float e = 0.f;
    const float tot = (float)BATCH;
    for (int i = threadIdx.x; i < NUM_LATENTS; i += 256) {
        unsigned c = counts[i];
        if (c) { float p = (float)c / tot; e -= p * logf(p); }
    }
    sh[threadIdx.x] = e;
    __syncthreads();
    for (int s = 128; s > 0; s >>= 1) {
        if ((int)threadIdx.x < s) sh[threadIdx.x] += sh[threadIdx.x + s];
        __syncthreads();
    }
    if (threadIdx.x == 0) {
        float H   = sh[0];
        float mse = *sqacc / (float)((size_t)BATCH * DIM);
        float el  = -H;
        float vq  = mse + 0.25f * mse + 0.1f * el;
        size_t base = (size_t)BATCH * DIM;
        out[base + 0] = vq;
        out[base + 1] = mse;   // q_loss
        out[base + 2] = mse;   // commit_loss
        out[base + 3] = el;    // entropy_loss
        out[base + 4] = H;     // entropy
    }
}

extern "C" void kernel_launch(void* const* d_in, const int* in_sizes, int n_in,
                              void* d_out, int out_size, void* d_ws, size_t ws_size,
                              hipStream_t stream) {
    const float* x  = (const float*)d_in[0];
    const float* cb = (const float*)d_in[1];
    char* ws = (char*)d_ws;
    unsigned* frags  = (unsigned*)ws;                         // 2 MB B-fragments
    float*    invn   = (float*)(ws + 2097152);                // 16 KB
    unsigned* counts = (unsigned*)(ws + 2097152 + 16384);     // 16 KB
    float*    sqacc  = (float*)(ws + 2097152 + 32768);        // 4 B
    float*    out    = (float*)d_out;

    vq_prep  <<<16,   256, 0, stream>>>(cb, invn, counts, sqacc);
    vq_pack_b<<<2048, 256, 0, stream>>>(cb, invn, frags);
    vq_main  <<<1024,  32, 0, stream>>>(x, cb, frags, counts, sqacc, out);
    vq_final <<<1,    256, 0, stream>>>(counts, sqacc, out);
}
